// BernMLPAugmenter_83640193122891
// MI455X (gfx1250) — matrix operand order repaired and checked by reading, not compiled, not run on hardware
//
#include <hip/hip_runtime.h>
#include <cmath>

typedef __attribute__((ext_vector_type(16))) _Float16 v16h;
typedef __attribute__((ext_vector_type(8)))  _Float16 v8h;
typedef __attribute__((ext_vector_type(8)))  float    v8f;
typedef __attribute__((ext_vector_type(4)))  float    v4f;

#define EMB   128
#define HID   64
#define WAVES 8
#define HPAD  68   // 16x64 h-tile row stride (floats), padded for LDS banks

// ---- pre-pass: node_emb f32 -> f16 table in d_ws (4 elems / thread) ----
__global__ __launch_bounds__(256) void cvt_emb_f16(
    const float* __restrict__ in, _Float16* __restrict__ out, int n)
{
    int i = (blockIdx.x * blockDim.x + threadIdx.x) * 4;
    if (i + 3 < n) {
        const v4f x = *(const v4f*)(in + i);
        out[i + 0] = (_Float16)x.x;
        out[i + 1] = (_Float16)x.y;
        out[i + 2] = (_Float16)x.z;
        out[i + 3] = (_Float16)x.w;
    } else {
        for (; i < n; ++i) out[i] = (_Float16)in[i];
    }
}

// One wave handles one tile of 16 edges:
//   A (16 x 256 f16, gathered node embeddings) x B (256 x 64 = W1 as f16)
//   -> C (16 x 64 f32) via 8 K-steps x 4 N-tiles of v_wmma_f32_16x16x32_f16.
// Then relu(C + b1) -> LDS, 64-dot with W2, sigmoid gate, 3 stores.
template <bool USE_F16>
__global__ __launch_bounds__(256) void bern_mlp_wmma(
    const void*  __restrict__ emb_v,       // f16 table (d_ws) or f32 node_emb
    const int*   __restrict__ edge_index,
    const float* __restrict__ eps_raw,
    const float* __restrict__ W1,
    const float* __restrict__ b1,
    const float* __restrict__ W2,
    const float* __restrict__ b2,
    float*       __restrict__ out,
    int M, int nTiles, int edge_stride)
{
    // B fragments of W1, pre-swizzled to the WMMA 16-bit B layout:
    // [kb(8)][nt(4)][lane(32)][16 f16]
    __shared__ alignas(32) _Float16 sB[8 * 4 * 32 * 16];   // 32 KB
    __shared__ alignas(16) float sH[WAVES][16][HPAD];       // ~34 KB h staging
    __shared__ float sW2[HID];
    __shared__ float sB1[HID];

    const int t = threadIdx.x;

    // ---- one-time per block: W1 -> f16 B fragments in LDS ----
    for (int trip = t; trip < 8 * 4 * 32; trip += 256) {
        const int lane  = trip & 31;
        const int nt    = (trip >> 5) & 3;
        const int kb    = trip >> 7;
        const int kbase = kb * 32 + ((lane & 16) ? 8 : 0);
        const int col   = nt * 16 + (lane & 15);
        _Float16* dst = &sB[trip * 16];
        #pragma unroll
        for (int d = 0; d < 8; ++d) {
            dst[2 * d + 0] = (_Float16)W1[(kbase + 2 * d + 0 + (d >= 4 ? 8 : 0)) * HID + col];
            dst[2 * d + 1] = (_Float16)W1[(kbase + 2 * d + 1 + (d >= 4 ? 8 : 0)) * HID + col];
        }
    }
    if (t < HID) { sW2[t] = W2[t]; sB1[t] = b1[t]; }
    __syncthreads();

    const int  wave  = t >> 5;
    const int  lane  = t & 31;
    const int  m     = lane & 15;           // edge row within tile
    const bool hi    = (lane & 16) != 0;    // high half of wave
    const float bias2 = b2[0];

    for (int tile = blockIdx.x * WAVES + wave; tile < nTiles;
         tile += gridDim.x * WAVES) {

        const int e  = tile * 16 + m;
        const int ec = (e < M) ? e : (M - 1);
        const int srcn = edge_index[ec];
        const int dstn = edge_index[edge_stride + ec];

        v8f acc[4] = {};   // 4 N-tiles of 16x16 f32 accumulators

        #pragma unroll
        for (int kb = 0; kb < 8; ++kb) {
            // K chunk kb*32..kb*32+31: first 128 features = src node, rest = dst.
            const int node = (kb < 4) ? srcn : dstn;
            const int fsub = ((kb * 32) & 127) + (hi ? 8 : 0);
            // A layout: low lanes hold K {0..7, 16..23}; high lanes {8..15, 24..31}
            v16h a;
            if constexpr (USE_F16) {
                const _Float16* rp =
                    (const _Float16*)emb_v + (size_t)node * EMB + fsub;
                const v8h lo = *(const v8h*)(rp);        // 16 B
                const v8h hh = *(const v8h*)(rp + 16);   // 16 B
                #pragma unroll
                for (int i = 0; i < 8; ++i) { a[i] = lo[i]; a[8 + i] = hh[i]; }
            } else {
                const float* rp = (const float*)emb_v + (size_t)node * EMB + fsub;
                const v4f x0 = *(const v4f*)(rp);
                const v4f x1 = *(const v4f*)(rp + 4);
                const v4f x2 = *(const v4f*)(rp + 16);
                const v4f x3 = *(const v4f*)(rp + 20);
                a[0]  = (_Float16)x0.x; a[1]  = (_Float16)x0.y;
                a[2]  = (_Float16)x0.z; a[3]  = (_Float16)x0.w;
                a[4]  = (_Float16)x1.x; a[5]  = (_Float16)x1.y;
                a[6]  = (_Float16)x1.z; a[7]  = (_Float16)x1.w;
                a[8]  = (_Float16)x2.x; a[9]  = (_Float16)x2.y;
                a[10] = (_Float16)x2.z; a[11] = (_Float16)x2.w;
                a[12] = (_Float16)x3.x; a[13] = (_Float16)x3.y;
                a[14] = (_Float16)x3.z; a[15] = (_Float16)x3.w;
            }

            const v16h* bp = (const v16h*)sB + (kb * 4) * 32 + lane;
            #pragma unroll
            for (int nt = 0; nt < 4; ++nt) {
                acc[nt] = __builtin_amdgcn_wmma_f32_16x16x32_f16(
                    /*neg_a=*/false, a, /*neg_b=*/false, bp[nt * 32],
                    /*c_mod=*/(short)0, acc[nt],
                    /*reuse_a=*/false, /*reuse_b=*/false);
            }
        }

        // ---- epilogue: h = relu(C + b1) -> LDS (per documented C layout) ----
        const int mrow = hi ? 8 : 0;   // lanes 16-31 hold M = r + 8
        #pragma unroll
        for (int nt = 0; nt < 4; ++nt) {
            const int n  = nt * 16 + m;   // column = lane % 16 within N-tile
            const float bb = sB1[n];
            #pragma unroll
            for (int r = 0; r < 8; ++r) {
                float h = acc[nt][r] + bb;
                h = h > 0.f ? h : 0.f;
                sH[wave][mrow + r][n] = h;
            }
        }
        // same-wave LDS store->load: DS ops are in-order; compiler inserts waits

        // ---- 64-dot with W2, split across lane halves ----
        float dot = 0.f;
        {
            const int nbase = hi ? 32 : 0;
            const float* hrow = &sH[wave][m][nbase];
            #pragma unroll
            for (int n = 0; n < 32; n += 4) {
                const v4f hv = *(const v4f*)(hrow + n);
                const v4f wv = *(const v4f*)(&sW2[nbase + n]);
                dot += hv.x * wv.x + hv.y * wv.y + hv.z * wv.z + hv.w * wv.w;
            }
        }
        dot += __shfl_xor(dot, 16, 32);

        if (!hi && e < M) {
            const float logit = dot + bias2;
            const float er    = eps_raw[e];
            const float bias  = 1e-4f;
            const float eps   = (2.f * bias - 1.f) * er + (1.f - bias);
            const float gate  = __logf(eps) - log1pf(-eps);
            const float w     = 1.f / (1.f + __expf(-(gate + logit)));
            out[e]         = w;  // sym_vals[:M]
            out[e + M]     = w;  // sym_vals[M:2M]
            out[e + 2 * M] = w;  // aug_edge_weight
        }
    }
}

extern "C" void kernel_launch(void* const* d_in, const int* in_sizes, int n_in,
                              void* d_out, int out_size, void* d_ws, size_t ws_size,
                              hipStream_t stream) {
    const float* node_emb   = (const float*)d_in[0];
    const int*   edge_index = (const int*)d_in[1];
    const float* eps_raw    = (const float*)d_in[2];
    const float* W1         = (const float*)d_in[3];
    const float* b1         = (const float*)d_in[4];
    const float* W2         = (const float*)d_in[5];
    const float* b2         = (const float*)d_in[6];
    float* out = (float*)d_out;

    const int nEmb = in_sizes[0];            // N_NODES * 128
    const int M = in_sizes[2];               // eps_raw has M elements
    const int edge_stride = in_sizes[1] / 2; // edge_index is (2, N_EDGES) flat
    const int nTiles = (M + 15) / 16;

    int blocks = (nTiles + WAVES - 1) / WAVES;
    if (blocks > 2048) blocks = 2048;

    const bool useF16 = (ws_size >= (size_t)nEmb * sizeof(_Float16));

    if (useF16) {
        _Float16* emb16 = (_Float16*)d_ws;
        const int cvtThreads = (nEmb + 3) / 4;
        const int cvtBlocks  = (cvtThreads + 255) / 256;
        hipLaunchKernelGGL(cvt_emb_f16, dim3(cvtBlocks), dim3(256), 0, stream,
                           node_emb, emb16, nEmb);
        hipLaunchKernelGGL((bern_mlp_wmma<true>), dim3(blocks), dim3(256), 0,
                           stream, (const void*)emb16, edge_index, eps_raw,
                           W1, b1, W2, b2, out, M, nTiles, edge_stride);
    } else {
        hipLaunchKernelGGL((bern_mlp_wmma<false>), dim3(blocks), dim3(256), 0,
                           stream, (const void*)node_emb, edge_index, eps_raw,
                           W1, b1, W2, b2, out, M, nTiles, edge_stride);
    }
}
